// Sampler_10213432230547
// MI455X (gfx1250) — matrix operand order repaired and checked by reading, not compiled, not run on hardware
//
#include <hip/hip_runtime.h>
#include <cstdint>

// ---- reference constants ----
#define TOPP          0.9f
#define LW_TOPK       10
#define NUM_LOGPROBS  20
#define EOS_TOKEN_ID  128001
#define EOS_POSITION  150
#define THRESH_BASE   (-0.71f)
#define THRESH_COEFF  0.001f
#define SAMPLING_EPS  1e-5f

// ---- tuning ----
#define NBIN        2048      // 11-bit monotonic-float-key histogram (sign+exp+2 mantissa bits)
#define CAP         512       // per-row candidate buffer (threshold keeps survivors ~topk+~40)
#define P1_THREADS  256       // 8 waves / block
#define TILE_ELEMS  1024      // P1_THREADS * 4 floats per async tile (4KB)

typedef __attribute__((ext_vector_type(16))) _Float16 v16h;
typedef __attribute__((ext_vector_type(8)))  float    v8f;

// monotonic float <-> uint key (ascending)
__device__ __forceinline__ unsigned f2key(float x) {
  unsigned b = __float_as_uint(x);
  return (b & 0x80000000u) ? ~b : (b | 0x80000000u);
}
__device__ __forceinline__ float key2f(unsigned k) {
  unsigned b = (k & 0x80000000u) ? (k ^ 0x80000000u) : ~k;
  return __uint_as_float(b);
}

// ---------------------------------------------------------------------------
// Pass 1: one block per row. Streams the row through LDS using the gfx1250
// async-to-LDS path (ASYNCcnt, double-buffered), computing online softmax
// (rowmax, sum exp) and a 2048-bin key histogram used to select a threshold
// bin whose suffix population >= top_k.
// ---------------------------------------------------------------------------
__global__ void __launch_bounds__(P1_THREADS)
sampler_pass1(const float* __restrict__ logits, int V,
              float* __restrict__ rowmax, float* __restrict__ rowsum,
              unsigned* __restrict__ hist, int* __restrict__ thrbin,
              unsigned* __restrict__ candcnt, const int* __restrict__ topk_p)
{
  const int row = blockIdx.x;
  const int tid = threadIdx.x;

  __shared__ __align__(16) float sbuf[2][TILE_ELEMS];
  __shared__ unsigned shist[NBIN];
  __shared__ float sm[P1_THREADS], ss[P1_THREADS];

  for (int i = tid; i < NBIN; i += P1_THREADS) shist[i] = 0u;
  __syncthreads();

  const float* base = logits + (size_t)row * (size_t)V;
  const unsigned long long base64 = (unsigned long long)(uintptr_t)base;
  // low 32 bits of a generic LDS pointer are the LDS byte offset (aperture map)
  const unsigned lds_base = (unsigned)(uintptr_t)(&sbuf[0][0]);

  float m = -INFINITY, s = 0.f;
  const int nit = V / TILE_ELEMS;

  // prologue: async-copy tile 0 into buffer 0  (gfx1250 ASYNCcnt path)
  {
    unsigned voff  = (unsigned)(tid * 16);
    unsigned laddr = lds_base + (unsigned)(tid * 16);
    asm volatile("global_load_async_to_lds_b128 %0, %1, %2"
                 :: "v"(laddr), "v"(voff), "s"(base64) : "memory");
  }
  for (int it = 0; it < nit; ++it) {
    const int bi = it & 1;
    if (it + 1 < nit) {
      unsigned voff  = (unsigned)(((it + 1) * TILE_ELEMS + tid * 4) * 4);
      unsigned laddr = lds_base + (unsigned)((bi ^ 1) * (TILE_ELEMS * 4) + tid * 16);
      asm volatile("global_load_async_to_lds_b128 %0, %1, %2"
                   :: "v"(laddr), "v"(voff), "s"(base64) : "memory");
      asm volatile("s_wait_asynccnt 1" ::: "memory");   // oldest (current tile) done
    } else {
      asm volatile("s_wait_asynccnt 0" ::: "memory");
    }
    const float4 v = *reinterpret_cast<const float4*>(&sbuf[bi][tid * 4]);
    float xs[4] = {v.x, v.y, v.z, v.w};
#pragma unroll
    for (int j = 0; j < 4; ++j) {
      const float x = xs[j];
      atomicAdd(&shist[f2key(x) >> 21], 1u);
      const float nm = fmaxf(m, x);                 // branchless online softmax
      s = s * __expf(m - nm) + __expf(x - nm);
      m = nm;
    }
  }
  // tail (dead for V=128000, kept for generality)
  for (int e = nit * TILE_ELEMS + tid; e < V; e += P1_THREADS) {
    const float x = base[e];
    atomicAdd(&shist[f2key(x) >> 21], 1u);
    const float nm = fmaxf(m, x);
    s = s * __expf(m - nm) + __expf(x - nm);
    m = nm;
  }

  sm[tid] = m; ss[tid] = s;
  __syncthreads();
  for (int off = P1_THREADS / 2; off > 0; off >>= 1) {
    if (tid < off) {
      float m2 = sm[tid + off], s2 = ss[tid + off];
      float M  = fmaxf(sm[tid], m2);
      ss[tid]  = ss[tid] * __expf(sm[tid] - M) + s2 * __expf(m2 - M);
      sm[tid]  = M;
    }
    __syncthreads();
  }
  for (int i = tid; i < NBIN; i += P1_THREADS)
    hist[(size_t)row * NBIN + i] = shist[i];
  if (tid == 0) {
    rowmax[row]  = sm[0];
    rowsum[row]  = ss[0];
    candcnt[row] = 0u;
    const int need = *topk_p;
    unsigned acc = 0u; int b = 0;
    for (int i = NBIN - 1; i >= 0; --i) {
      acc += shist[i];
      if ((int)acc >= need) { b = i; break; }
    }
    thrbin[row] = b;
  }
}

// ---------------------------------------------------------------------------
// Pass 2: gather candidates >= threshold bin into a small per-row buffer.
// L2-resident pass (131MB tensor fits the 192MB L2). Uses global_prefetch.
// Each candidate packs (key, ~index): descending u64 sort => value-desc,
// index-asc tie-break, matching jnp.argmax / lax.top_k tie rules.
// ---------------------------------------------------------------------------
__global__ void __launch_bounds__(256)
sampler_pass2(const float* __restrict__ logits, int V,
              const int* __restrict__ thrbin,
              unsigned* __restrict__ candcnt,
              unsigned long long* __restrict__ cand)
{
  const int row = blockIdx.x;
  const int tid = threadIdx.x;
  const unsigned thr = ((unsigned)thrbin[row]) << 21;
  const float* base = logits + (size_t)row * (size_t)V;

  for (int e = tid * 4; e + 3 < V; e += 1024) {
    __builtin_prefetch(base + e + 16384, 0, 0);   // global_prefetch_b8
    const float4 v = *reinterpret_cast<const float4*>(base + e);
    float xs[4] = {v.x, v.y, v.z, v.w};
#pragma unroll
    for (int j = 0; j < 4; ++j) {
      const unsigned key = f2key(xs[j]);
      if (key >= thr) {
        const unsigned pos = atomicAdd(&candcnt[row], 1u);
        if (pos < CAP)
          cand[(size_t)row * CAP + pos] =
              ((unsigned long long)key << 32) | (unsigned)~(unsigned)(e + j);
      }
    }
  }
  for (int e = (V & ~3) + tid; e < V; e += 256) {   // tail (dead for V=128000)
    const unsigned key = f2key(base[e]);
    if (key >= thr) {
      const unsigned pos = atomicAdd(&candcnt[row], 1u);
      if (pos < CAP)
        cand[(size_t)row * CAP + pos] =
            ((unsigned long long)key << 32) | (unsigned)~(unsigned)e;
    }
  }
}

// ---------------------------------------------------------------------------
// Pass 3: one block (4 waves) per row. Bitonic-sort candidates in LDS,
// run the sampler logic on the top-k set. The nucleus cumulative sums are
// computed with V_WMMA_F32_16X16X32_F16: A = 16x32 lower-triangular ones
// (K<16), B columns = 16-element probability chunks => D[m,n] = inclusive
// intra-chunk prefix sums (f32 accumulate), combined with chunk totals.
// ---------------------------------------------------------------------------
__global__ void __launch_bounds__(128)
sampler_finalize(const float* __restrict__ logits, int V, int B,
                 const float* __restrict__ temperature,
                 const int* __restrict__ token_lengths,
                 const int* __restrict__ output_lengths,
                 const float* __restrict__ rowmax, const float* __restrict__ rowsum,
                 const unsigned* __restrict__ hist, const int* __restrict__ thrbin,
                 const unsigned* __restrict__ candcnt,
                 const unsigned long long* __restrict__ cand,
                 const int* __restrict__ topk_p,
                 float* __restrict__ out)
{
  const int row = blockIdx.x;
  const int tid = threadIdx.x;

  __shared__ unsigned long long s_cand[CAP];
  __shared__ __align__(16) float s_pf[256];   // WMMA D: intra-chunk prefixes
  __shared__ float s_q[256];     // probabilities (chunk layout source for B)
  __shared__ float s_P[64];      // global inclusive prefix sums
  __shared__ float s_val[64];
  __shared__ int   s_idx[64];
  __shared__ float s_f[2];       // [0]=Z (top-k softmax denom), [1]=tok_logit
  __shared__ int   s_i[1];       // [0]=sampled
  __shared__ unsigned s_rank;

  const int   n    = min((int)candcnt[row], CAP);
  const float temp = temperature[row];
  const int   topk = min(min(*topk_p, 64), (n > 0 ? n : 1));

  for (int i = tid; i < CAP; i += blockDim.x)
    s_cand[i] = (i < n) ? cand[(size_t)row * CAP + i] : 0ull;
  __syncthreads();

  // bitonic sort, descending (value desc, index asc on ties)
  for (int k = 2; k <= CAP; k <<= 1) {
    for (int j = k >> 1; j > 0; j >>= 1) {
      for (int i = tid; i < CAP; i += blockDim.x) {
        const int ixj = i ^ j;
        if (ixj > i) {
          const unsigned long long a = s_cand[i], b = s_cand[ixj];
          const bool desc = ((i & k) == 0);
          if ((a < b) == desc) { s_cand[i] = b; s_cand[ixj] = a; }
        }
      }
      __syncthreads();
    }
  }

  if (tid < 64) {
    if (tid < n) {
      const unsigned long long c = s_cand[tid];
      s_val[tid] = key2f((unsigned)(c >> 32));
      s_idx[tid] = (int)~(unsigned)(c & 0xffffffffu);
    } else { s_val[tid] = -INFINITY; s_idx[tid] = 0; }
  }
  __syncthreads();

  // top-k softmax of temperature-scaled logits: e_j = exp(s_j - s_0)
  const float s0 = s_val[0] / temp;
  for (int i = tid; i < 256; i += blockDim.x) {
    float e = 0.f;
    if (i < topk) e = __expf(s_val[i] / temp - s0);
    s_q[i] = e;
  }
  __syncthreads();
  if (tid == 0) {
    float z = 0.f;
    for (int i = 0; i < topk; ++i) z += s_q[i];
    s_f[0] = z;
  }
  __syncthreads();
  const float Z = s_f[0];
  for (int i = tid; i < 64; i += blockDim.x) s_q[i] = s_q[i] / Z;
  __syncthreads();

  // ---- WMMA prefix scan: D = L(16x32 tri) x Q(32x16), f32 accumulate ----
  // Branch-free operand construction: for lane L, A rows M = L&15 with
  // K-halves starting at kbase = (L>>4)*8; B columns N = L&15 hold rows
  // K=kbase..kbase+7 in b[0..7] and zeros in b[8..15] (K>=16).
  {
    const int lane  = tid & 31;
    const int M     = lane & 15;
    const int kbase = (lane >> 4) << 3;
    const bool lowhalf = (lane < 16);
    v16h a, b;
#pragma unroll
    for (int j = 0; j < 8; ++j) {
      a[j]     = (_Float16)(((kbase + j) <= M) ? 1.f : 0.f);  // K = kbase+j
      a[8 + j] = (_Float16)0.f;                               // K = kbase+16+j >= 16
    }
#pragma unroll
    for (int j = 0; j < 16; ++j)
      b[j] = (_Float16)(j < 8 ? s_q[M * 16 + kbase + j] : 0.f);     // rows K<16 only
    v8f d = {};
    d = __builtin_amdgcn_wmma_f32_16x16x32_f16(
            false, a, false, b, (short)0, d, false, false);
    if (tid < 32) {   // wave 0 stores; D lane layout -> contiguous 8 floats
      const int pfbase = M * 16 + kbase;   // n=(lane&15), m base=(lane>=16)*8
      *reinterpret_cast<float4*>(&s_pf[pfbase])     = make_float4(d[0], d[1], d[2], d[3]);
      *reinterpret_cast<float4*>(&s_pf[pfbase + 4]) = make_float4(d[4], d[5], d[6], d[7]);
    }
  }
  __syncthreads();
  if (tid < 64) {                       // global inclusive prefix = intra + chunk offsets
    const int c = tid >> 4, mm = tid & 15;
    float off = 0.f;
    for (int cc = 0; cc < c; ++cc) off += s_pf[cc * 16 + 15];
    s_P[tid] = s_pf[c * 16 + mm] + off;
  }
  __syncthreads();

  if (tid == 0) {
    // nucleus: keep descending rank r iff exclusive prefix < TOP_P (r=0 always)
    int K = 1;
    while (K < topk && s_P[K - 1] < TOPP) ++K;
    const float mass   = s_P[K - 1];
    const float max_lp = -(__logf(Z) + __logf(mass));   // max of filtered log_softmax

    // top-10 of filtered: kept ranks first, then smallest masked token ids
    int tk[LW_TOPK];
    int filled = 0;
    for (int r = 0; r < K && filled < LW_TOPK; ++r) tk[filled++] = s_idx[r];
    for (int t = 0; filled < LW_TOPK; ++t) {
      bool used = false;
      for (int r = 0; r < K; ++r) if (s_idx[r] == t) { used = true; break; }
      if (!used) tk[filled++] = t;
    }
    int best = 0, bestlen = -1;
    for (int i = 0; i < LW_TOPK; ++i) {
      int ti = tk[i]; ti = ti < 0 ? 0 : (ti >= V ? V - 1 : ti);
      const int len = token_lengths[ti];
      if (len > bestlen) { bestlen = len; best = i; }   // first max, like jnp.argmax
    }
    int rs = tk[best];

    const int   olen   = output_lengths[row];
    const float thr_lp = THRESH_BASE + THRESH_COEFF * (float)olen;
    if (olen >= EOS_POSITION && max_lp < thr_lp) rs = EOS_TOKEN_ID;

    const int sampled = (temp < SAMPLING_EPS) ? s_idx[0] : rs;
    s_i[0] = sampled;
    const int cidx = sampled < (V - 1) ? sampled : (V - 1);   // JAX gather clip
    s_f[1]  = logits[(size_t)row * V + cidx];
    s_rank  = 0u;
  }
  __syncthreads();

  // rank = count(logits >= tok_logit): exact from candidates when tok above the
  // selection threshold (all such elements are candidates); histogram suffix
  // otherwise (forced-EOS clipped-gather corner).
  const float    tok_logit = s_f[1];
  const unsigned tkey = f2key(tok_logit);
  const int      tbin = (int)(tkey >> 21);
  const int      tb   = thrbin[row];
  unsigned local = 0u;
  if (tbin >= tb) {
    for (int i = tid; i < n; i += blockDim.x)
      if ((unsigned)(s_cand[i] >> 32) >= tkey) ++local;
  } else {
    for (int i = tbin + tid; i < NBIN; i += blockDim.x)
      local += hist[(size_t)row * NBIN + i];
  }
  if (local) atomicAdd(&s_rank, local);
  __syncthreads();

  if (tid == 0) {
    float* out_sampled = out;                       // [B]
    float* out_idx     = out + B;                   // [B,21]
    float* out_lps     = out + B + B * 21;          // [B,21]
    float* out_rank    = out + B + 2 * B * 21;      // [B]
    const float rmax   = rowmax[row];
    const float logZ   = __logf(rowsum[row]);
    const int sampled  = s_i[0];
    const float tok_lp = tok_logit - rmax - logZ;
    out_sampled[row] = (float)sampled;
    out_idx[(size_t)row * 21 + 0] = (float)sampled;
    out_lps[(size_t)row * 21 + 0] = tok_lp;
    for (int i = 0; i < NUM_LOGPROBS; ++i) {
      out_idx[(size_t)row * 21 + 1 + i] = (float)s_idx[i];
      out_lps[(size_t)row * 21 + 1 + i] = s_val[i] - rmax - logZ;
    }
    out_rank[row] = (float)s_rank;
  }
}

// ---------------------------------------------------------------------------
extern "C" void kernel_launch(void* const* d_in, const int* in_sizes, int n_in,
                              void* d_out, int out_size, void* d_ws, size_t ws_size,
                              hipStream_t stream) {
  const float* logits         = (const float*)d_in[0];
  const float* temperature    = (const float*)d_in[1];
  const int*   token_lengths  = (const int*)d_in[2];
  const int*   output_lengths = (const int*)d_in[3];
  const int*   topk_p         = (const int*)d_in[4];   // device scalar
  const int B = in_sizes[1];                            // 256
  const int V = in_sizes[2];                            // 128000
  float* out = (float*)d_out;

  char* w = (char*)d_ws;
  float*    rowmax  = (float*)w;    w += (size_t)B * 4;
  float*    rowsum  = (float*)w;    w += (size_t)B * 4;
  int*      thrbin  = (int*)w;      w += (size_t)B * 4;
  unsigned* candcnt = (unsigned*)w; w += (size_t)B * 4;
  unsigned long long* cand = (unsigned long long*)w; w += (size_t)B * CAP * 8;
  unsigned* hist    = (unsigned*)w; w += (size_t)B * NBIN * 4;
  (void)ws_size; (void)n_in; (void)out_size;

  sampler_pass1<<<B, P1_THREADS, 0, stream>>>(logits, V, rowmax, rowsum, hist,
                                              thrbin, candcnt, topk_p);
  sampler_pass2<<<B, 256, 0, stream>>>(logits, V, thrbin, candcnt, cand);
  sampler_finalize<<<B, 128, 0, stream>>>(logits, V, B, temperature,
                                          token_lengths, output_lengths,
                                          rowmax, rowsum, hist, thrbin,
                                          candcnt, cand, topk_p, out);
}
